// InfoNCELossFG_16758962389001
// MI455X (gfx1250) — matrix operand order repaired and checked by reading, not compiled, not run on hardware
//
#include <hip/hip_runtime.h>
#include <math.h>

typedef __attribute__((ext_vector_type(16))) _Float16 v16h;
typedef __attribute__((ext_vector_type(8)))  float    v8f;
typedef __attribute__((ext_vector_type(4)))  float    f32x4;

#define B_  1024
#define N_  256
#define D_  512

// ws layout (floats): [0]=pos exp-sum accum, [1]=bg exp-sum accum, [2..2+B) = 1/||img_b||
__global__ void infonce_zero_kernel(float* __restrict__ ws) {
    if (threadIdx.x < 2) ws[threadIdx.x] = 0.0f;
}

// One block per batch row: inv-norm of img row + exp(pos logit) accumulation.
__global__ __launch_bounds__(256) void infonce_pos_kernel(const float* __restrict__ img,
                                                          const float* __restrict__ pro,
                                                          float* __restrict__ ws) {
    __shared__ float s2[256];
    __shared__ float sd[256];
    const int b = blockIdx.x;
    const int t = threadIdx.x;
    const float* irow = img + (size_t)b * D_;
    const float* prow = pro + (size_t)b * D_;
    float x0 = irow[t], x1 = irow[t + 256];
    float p0 = prow[t], p1 = prow[t + 256];
    s2[t] = x0 * x0 + x1 * x1;
    sd[t] = x0 * p0 + x1 * p1;
    __syncthreads();
    for (int s = 128; s > 0; s >>= 1) {
        if (t < s) { s2[t] += s2[t + s]; sd[t] += sd[t + s]; }
        __syncthreads();
    }
    if (t == 0) {
        float inv = rsqrtf(s2[0]);
        ws[2 + b] = inv;                      // consumed by bg kernel (later in stream)
        atomicAdd(&ws[0], expf(sd[0] * inv)); // TEMPERATURE = 1.0
    }
}

// Main HBM-bound kernel: one wave per (b, 16-wide n tile); 16x16x32 f16 WMMA over K=512.
__global__ __launch_bounds__(256) void infonce_bg_kernel(const float* __restrict__ img,
                                                         const float* __restrict__ bg,
                                                         float* __restrict__ ws) {
    __shared__ float red[16];
    const int lane = threadIdx.x & 31;
    const int wave = threadIdx.x >> 5;
    const int wid  = blockIdx.x * 8 + wave;   // 16384 waves total
    const int b    = wid >> 4;                // 0..1023
    const int nt   = wid & 15;                // 0..15 -> n tile of 16
    const int m    = lane & 15;               // A-matrix row within tile
    const int kh   = lane >> 4;               // K-half select per ISA A/B layouts

    // A: row (b, nt*16+m) of bg; lane covers K = kbase + kh*8 + [0..7] and kbase + 16 + kh*8 + [0..7]
    const float* arow = bg + ((size_t)(b * N_ + nt * 16 + m)) * (size_t)D_ + kh * 8;
    // B: img chunk broadcast across all 16 columns; lane covers K = kbase + kh*8 + [0..7] and kbase + 16 + kh*8 + [0..7]
    const float* brow = img + (size_t)b * D_ + kh * 8;

    v8f acc = {};
    #pragma unroll 2
    for (int kb = 0; kb < D_; kb += 32) {
        f32x4 a0 = __builtin_nontemporal_load((const f32x4*)(arow + kb + 0));
        f32x4 a1 = __builtin_nontemporal_load((const f32x4*)(arow + kb + 4));
        f32x4 a2 = __builtin_nontemporal_load((const f32x4*)(arow + kb + 16));
        f32x4 a3 = __builtin_nontemporal_load((const f32x4*)(arow + kb + 20));
        f32x4 b0 = *(const f32x4*)(brow + kb + 0);
        f32x4 b1 = *(const f32x4*)(brow + kb + 4);
        f32x4 b2 = *(const f32x4*)(brow + kb + 16);
        f32x4 b3 = *(const f32x4*)(brow + kb + 20);

        v16h A, Bm;
        A[0]  = (_Float16)a0.x; A[1]  = (_Float16)a0.y; A[2]  = (_Float16)a0.z; A[3]  = (_Float16)a0.w;
        A[4]  = (_Float16)a1.x; A[5]  = (_Float16)a1.y; A[6]  = (_Float16)a1.z; A[7]  = (_Float16)a1.w;
        A[8]  = (_Float16)a2.x; A[9]  = (_Float16)a2.y; A[10] = (_Float16)a2.z; A[11] = (_Float16)a2.w;
        A[12] = (_Float16)a3.x; A[13] = (_Float16)a3.y; A[14] = (_Float16)a3.z; A[15] = (_Float16)a3.w;
        Bm[0]  = (_Float16)b0.x; Bm[1]  = (_Float16)b0.y; Bm[2]  = (_Float16)b0.z; Bm[3]  = (_Float16)b0.w;
        Bm[4]  = (_Float16)b1.x; Bm[5]  = (_Float16)b1.y; Bm[6]  = (_Float16)b1.z; Bm[7]  = (_Float16)b1.w;
        Bm[8]  = (_Float16)b2.x; Bm[9]  = (_Float16)b2.y; Bm[10] = (_Float16)b2.z; Bm[11] = (_Float16)b2.w;
        Bm[12] = (_Float16)b3.x; Bm[13] = (_Float16)b3.y; Bm[14] = (_Float16)b3.z; Bm[15] = (_Float16)b3.w;

        // D = A(16x32) x B(32x16) + C ; every column of D equals the 16 dot products
        acc = __builtin_amdgcn_wmma_f32_16x16x32_f16(false, A, false, Bm,
                                                     (short)0, acc, false, false);
    }

    // lanes 0..15: acc[r] = dot for tile row r ; lanes 16..31: acc[r] = dot for row 8+r
    const float scale = ws[2 + b];  // 1/||img_b|| (TEMPERATURE = 1.0)
    float s = 0.0f;
    #pragma unroll
    for (int r = 0; r < 8; ++r) s += expf(scale * acc[r]);

    if (m == 0) red[wave * 2 + kh] = s;  // lanes 0 and 16 of each wave
    __syncthreads();
    if (threadIdx.x == 0) {
        float tot = 0.0f;
        #pragma unroll
        for (int i = 0; i < 16; ++i) tot += red[i];
        atomicAdd(&ws[1], tot);
    }
}

__global__ void infonce_fin_kernel(const float* __restrict__ ws, float* __restrict__ out) {
    float pos = ws[0];
    float neg = ws[0] + ws[1];
    out[0] = -logf(pos / neg);
}

extern "C" void kernel_launch(void* const* d_in, const int* in_sizes, int n_in,
                              void* d_out, int out_size, void* d_ws, size_t ws_size,
                              hipStream_t stream) {
    const float* fg_img = (const float*)d_in[0];   // [B, D]
    const float* fg_pro = (const float*)d_in[1];   // [B, D]
    const float* bg_pro = (const float*)d_in[2];   // [B, N, D]
    float* ws  = (float*)d_ws;
    float* out = (float*)d_out;

    infonce_zero_kernel<<<1, 32, 0, stream>>>(ws);
    infonce_pos_kernel<<<B_, 256, 0, stream>>>(fg_img, fg_pro, ws);
    // 1024 b * 16 n-tiles = 16384 waves / 8 waves per block = 2048 blocks
    infonce_bg_kernel<<<(B_ * (N_ / 16)) / 8, 256, 0, stream>>>(fg_img, bg_pro, ws);
    infonce_fin_kernel<<<1, 1, 0, stream>>>(ws, out);
}
